// MTINN_81003083203443
// MI455X (gfx1250) — matrix operand order repaired and checked by reading, not compiled, not run on hardware
//
#include <hip/hip_runtime.h>
#include <hip/hip_bf16.h>

typedef __attribute__((ext_vector_type(16))) __bf16 v16bf;
typedef __attribute__((ext_vector_type(8)))  __bf16 v8bf;
typedef __attribute__((ext_vector_type(8)))  float  v8f;

#define SEQ   1024
#define BT    16
#define NGATE 512

// K order per cell (permuted; A and B share it so the dot product is unchanged):
//   shared : [h_s(0..127) | x(128..134) | pad ]                      -> 5 chunks of 32
//   a      : [s(0..127) | h_a(128..255) | pa(256..258) x(259..265)]  -> 9 chunks
//   v      : [s | h_v | pv(256..258) a_t(259..261) x(262..268)]      -> 9 chunks
//   p      : [s | h_p | pp(256..258) v_t(259..261) x(262..268)]      -> 9 chunks
#define OFF_S 0
#define OFF_A 81920       // 5*32*32*16
#define OFF_V 229376
#define OFF_P 376832
#define WSW_HALVES 524288

#if __has_builtin(__builtin_amdgcn_tanhf)
#define TANHF(v) __builtin_amdgcn_tanhf(v)
#else
#define TANHF(v) tanhf(v)
#endif
__device__ __forceinline__ float sigf(float v) { return 0.5f * TANHF(0.5f * v) + 0.5f; }

// ---------------------------------------------------------------------------
// Weights -> bf16, pre-swizzled into the WMMA B-operand lane layout.
// B element (k within 32-chunk, n): lane = n + 16*((k>>3)&1), idx = (k & 7) + 8*(k>>4).
// Block layout [cell][kc][ntile][lane][16] -> one 32B coalesced load per lane.
// ---------------------------------------------------------------------------
__global__ void prep_weights(const float* __restrict__ WihS, const float* __restrict__ WhhS,
                             const float* __restrict__ WihA, const float* __restrict__ WhhA,
                             const float* __restrict__ WihV, const float* __restrict__ WhhV,
                             const float* __restrict__ WihP, const float* __restrict__ WhhP,
                             __bf16* __restrict__ wsw)
{
    int e = blockIdx.x * 256 + threadIdx.x;
    if (e >= WSW_HALVES) return;
    int c, base;
    if (e < OFF_A)      { c = 0; base = OFF_S; }
    else if (e < OFF_V) { c = 1; base = OFF_A; }
    else if (e < OFF_P) { c = 2; base = OFF_V; }
    else                { c = 3; base = OFF_P; }
    int r  = e - base;
    int kc = r >> 14;
    int nt = (r >> 9) & 31;
    int l  = (r >> 4) & 31;
    int i  = r & 15;
    int n  = nt * 16 + (l & 15);
    int k  = kc * 32 + (i & 7) + 8 * (l >> 4) + 16 * (i >> 3);

    float v = 0.f;
    if (c == 0) {
        if (k < 128)      v = WhhS[n * 128 + k];
        else if (k < 135) v = WihS[n * 7 + (k - 128)];
    } else if (c == 1) {
        if (k < 128)      v = WihA[n * 138 + k];
        else if (k < 256) v = WhhA[n * 128 + (k - 128)];
        else if (k < 266) v = WihA[n * 138 + 128 + (k - 256)];
    } else if (c == 2) {
        if (k < 128)      v = WihV[n * 141 + k];
        else if (k < 256) v = WhhV[n * 128 + (k - 128)];
        else if (k < 269) v = WihV[n * 141 + 128 + (k - 256)];
    } else {
        if (k < 128)      v = WihP[n * 141 + k];
        else if (k < 256) v = WhhP[n * 128 + (k - 128)];
        else if (k < 269) v = WihP[n * 141 + 128 + (k - 256)];
    }
    wsw[e] = (__bf16)v;
}

__global__ void prep_bias(const float* __restrict__ b0i, const float* __restrict__ b0h,
                          const float* __restrict__ b1i, const float* __restrict__ b1h,
                          const float* __restrict__ b2i, const float* __restrict__ b2h,
                          const float* __restrict__ b3i, const float* __restrict__ b3h,
                          float* __restrict__ bsum)
{
    int e = blockIdx.x * 256 + threadIdx.x;
    if (e >= 4 * NGATE) return;
    int c = e >> 9, n = e & 511;
    const float* bi = c == 0 ? b0i : c == 1 ? b1i : c == 2 ? b2i : b3i;
    const float* bh = c == 0 ? b0h : c == 1 ? b1h : c == 2 ? b2h : b3h;
    bsum[e] = bi[n] + bh[n];
}

// ---------------------------------------------------------------------------
// One LSTM cell: gate GEMM with gate-aligned N-tiles (wave w owns i/f/g/o for
// hidden columns j = w*16+nb), elementwise in D-fragment registers, h written
// straight back to A-swizzled LDS for the next consumer.
// ---------------------------------------------------------------------------
template <int NK, bool WRITE_HFC>
__device__ __forceinline__ void run_cell(const __bf16* __restrict__ A0,  // chunks 0-3 (h_s)
                                         const __bf16* __restrict__ A1,  // chunks 4-7 (h_cell) / tail for shared
                                         const __bf16* __restrict__ A2,  // chunk 8 (tail)
                                         const __bf16* __restrict__ wb,  // weights + lane*16 (opaque per t)
                                         const int lane, const int w,
                                         const float* __restrict__ bias,
                                         float* __restrict__ creg,
                                         __bf16* __restrict__ hw,        // A-swizzled h dest (ping-pong)
                                         __bf16* __restrict__ hfc,       // row-major h for FC head
                                         const int hswz, const int j)
{
    v8f acc0 = {}, acc1 = {}, acc2 = {}, acc3 = {};
    #pragma unroll
    for (int kc = 0; kc < NK; ++kc) {
        const __bf16* ap = (kc < 4) ? (A0 + kc * 512)
                         : (kc < 8) ? (A1 + (kc - 4) * 512)
                                    : A2;
        v16bf a  = *(const v16bf*)(ap + lane * 16);
        v16bf b0 = *(const v16bf*)(wb + ((kc * 32 + w     ) << 9));
        v16bf b1 = *(const v16bf*)(wb + ((kc * 32 + w + 8 ) << 9));
        v16bf b2 = *(const v16bf*)(wb + ((kc * 32 + w + 16) << 9));
        v16bf b3 = *(const v16bf*)(wb + ((kc * 32 + w + 24) << 9));
        acc0 = __builtin_amdgcn_wmma_f32_16x16x32_bf16(false, a, false, b0, (short)0, acc0, false, false);
        acc1 = __builtin_amdgcn_wmma_f32_16x16x32_bf16(false, a, false, b1, (short)0, acc1, false, false);
        acc2 = __builtin_amdgcn_wmma_f32_16x16x32_bf16(false, a, false, b2, (short)0, acc2, false, false);
        acc3 = __builtin_amdgcn_wmma_f32_16x16x32_bf16(false, a, false, b3, (short)0, acc3, false, false);
    }
    const int m0 = 8 * (lane >> 4);   // D layout: lanes 0-15 -> M=r, lanes 16-31 -> M=r+8
    #pragma unroll
    for (int r = 0; r < 8; ++r) {
        float gi = acc0[r] + bias[0];
        float gf = acc1[r] + bias[1];
        float gg = acc2[r] + bias[2];
        float go = acc3[r] + bias[3];
        float c  = sigf(gf) * creg[r] + sigf(gi) * TANHF(gg);
        creg[r]  = c;
        float h  = sigf(go) * TANHF(c);
        int m = m0 + r;
        hw[hswz + m * 16] = (__bf16)h;                 // A-fragment swizzle position
        if (WRITE_HFC) hfc[m * 128 + j] = (__bf16)h;   // row-major copy for FC head
    }
}

__global__ __launch_bounds__(256, 1)
void lstm_main(const float* __restrict__ x,
               const __bf16* __restrict__ wsw,
               const float* __restrict__ bsum,
               const float* __restrict__ WfcA, const float* __restrict__ bfcA,
               const float* __restrict__ WfcV, const float* __restrict__ bfcV,
               const float* __restrict__ WfcP, const float* __restrict__ bfcP,
               float* __restrict__ out)
{
    __shared__ __bf16 Hs[2][4 * 512];       // h_s, A-swizzled, ping-pong by t parity
    __shared__ __bf16 Hc[3][2][4 * 512];    // h_a/h_v/h_p, A-swizzled, ping-pong
    __shared__ __bf16 Ts[512];              // shared-cell tail chunk (x + zero pad)
    __shared__ __bf16 Tc[3][512];           // cascade tail chunks (carries, cur outs, x)
    __shared__ __bf16 hfc[16 * 128];        // row-major h of current cascade cell

    const int tid  = threadIdx.x;
    const int lane = tid & 31;
    const int w    = tid >> 5;              // wave id == hidden-column block 0..7
    const int nb   = lane & 15;
    const int wgBase = blockIdx.x * BT;

    // zero persistent LDS state (h(-1)=0, head carries=0, static pad stays 0)
    for (int e = tid; e < 2 * 2048; e += 256) (&Hs[0][0])[e] = (__bf16)0.f;
    for (int e = tid; e < 6 * 2048; e += 256) (&Hc[0][0][0])[e] = (__bf16)0.f;
    for (int e = tid; e < 512;      e += 256) Ts[e] = (__bf16)0.f;
    for (int e = tid; e < 3 * 512;  e += 256) (&Tc[0][0])[e] = (__bf16)0.f;

    // x_t scatter into tail chunks (A-swizzle positions), done by 112 threads
    auto scatter_x = [&](int tt, int s) {       // s = 0..111
        int m = s / 7, kx = s - m * 7;
        __bf16 xb = (__bf16)x[(size_t)(wgBase + m) * SEQ * 7 + (size_t)tt * 7 + kx];
        Ts[m * 16 + kx] = xb;                                   // shared: k=128+kx
        { int kk = 3 + kx;                                      // a: extras kk=3..9
          Tc[0][(m + 16 * ((kk >> 3) & 1)) * 16 + (kk & 7)] = xb; }
        { int kk = 6 + kx;                                      // v/p: extras kk=6..12
          int pos = (m + 16 * ((kk >> 3) & 1)) * 16 + (kk & 7);
          Tc[1][pos] = xb; Tc[2][pos] = xb; }
    };

    // FC head: 128 threads = 16 rows x 8 lanes, 16-wide partial dot + xor-tree
    auto fc_head = [&](const float* __restrict__ W, const float* __restrict__ bb,
                       __bf16* tcur, __bf16* tnext, int headOff, int tt) {
        const int mg = tid >> 3, g = tid & 7;   // row, lane-in-group
        v8bf h0 = *(const v8bf*)(hfc + mg * 128 + g * 16);
        v8bf h1 = *(const v8bf*)(hfc + mg * 128 + g * 16 + 8);
        float hv[16];
        #pragma unroll
        for (int u = 0; u < 8; ++u) { hv[u] = (float)h0[u]; hv[8 + u] = (float)h1[u]; }
        float s0 = 0.f, s1 = 0.f, s2 = 0.f;
        const float* w0 = W + g * 16;
        #pragma unroll
        for (int u = 0; u < 16; ++u) {
            s0 += hv[u] * w0[u];
            s1 += hv[u] * w0[128 + u];
            s2 += hv[u] * w0[256 + u];
        }
        #pragma unroll
        for (int off = 1; off < 8; off <<= 1) {
            s0 += __shfl_xor(s0, off);
            s1 += __shfl_xor(s1, off);
            s2 += __shfl_xor(s2, off);
        }
        if (g == 0) {
            float sv[3] = { s0 + bb[0], s1 + bb[1], s2 + bb[2] };
            #pragma unroll
            for (int oi = 0; oi < 3; ++oi) {
                if (tcur) tcur[mg * 16 + 3 + oi] = (__bf16)sv[oi];  // feed next cell this step
                tnext[mg * 16 + oi] = (__bf16)sv[oi];               // carry for next step
                out[(size_t)(wgBase + mg) * SEQ * 9 + (size_t)tt * 9 + headOff + oi] = sv[oi];
            }
        }
    };

    // per-lane constants: hidden column j, its A-swizzle position pieces
    const int j    = w * 16 + nb;
    const int kkj  = j & 31;
    const int parj = (kkj >> 3) & 1;
    const int iij  = (kkj & 7) + 8 * (kkj >> 4);
    const int hswz = (j >> 5) * 512 + 256 * parj + iij;

    // preload gate biases (t-invariant): gate g columns n = g*128 + j
    float biasv[4][4];
    #pragma unroll
    for (int ci = 0; ci < 4; ++ci)
        #pragma unroll
        for (int g = 0; g < 4; ++g)
            biasv[ci][g] = bsum[ci * NGATE + g * 128 + j];

    float creg[4][8];                       // c-state in D-fragment layout
    #pragma unroll
    for (int ci = 0; ci < 4; ++ci)
        #pragma unroll
        for (int r = 0; r < 8; ++r) creg[ci][r] = 0.f;

    if (tid < 112) scatter_x(0, tid);       // x_0
    __syncthreads();

    for (int t = 0; t < SEQ; ++t) {
        // opaque zero: keeps weight addresses loop-variant so the B-fragment
        // global loads stay in-loop (L2 hits) instead of being hoisted+spilled
        int nh = 0;
        asm volatile("" : "+s"(nh));
        const __bf16* wl = wsw + nh + (size_t)lane * 16;

        const int p = t & 1, q = p ^ 1;

        // ---- shared cell: reads h_s(t-1) from Hs[q], writes h_s(t) to Hs[p] ----
        run_cell<5, false>(&Hs[q][0], Ts, Ts, wl + OFF_S, lane, w,
                           biasv[0], creg[0], &Hs[p][0], hfc, hswz, j);
        __syncthreads();

        // ---- cell a ----
        run_cell<9, true>(&Hs[p][0], &Hc[0][q][0], Tc[0], wl + OFF_A, lane, w,
                          biasv[1], creg[1], &Hc[0][p][0], hfc, hswz, j);
        __syncthreads();
        if (tid < 128) fc_head(WfcA, bfcA, &Tc[1][0], &Tc[0][0], 0, t);
        __syncthreads();

        // ---- cell v ----
        run_cell<9, true>(&Hs[p][0], &Hc[1][q][0], Tc[1], wl + OFF_V, lane, w,
                          biasv[2], creg[2], &Hc[1][p][0], hfc, hswz, j);
        __syncthreads();
        if (tid < 128) fc_head(WfcV, bfcV, &Tc[2][0], &Tc[1][0], 3, t);
        __syncthreads();

        // ---- cell p ----
        run_cell<9, true>(&Hs[p][0], &Hc[2][q][0], Tc[2], wl + OFF_P, lane, w,
                          biasv[3], creg[3], &Hc[2][p][0], hfc, hswz, j);
        __syncthreads();
        // FC_p on waves 0-3; x_{t+1} scatter on waves 4-7 (disjoint LDS slots)
        if (tid < 128) fc_head(WfcP, bfcP, nullptr, &Tc[2][0], 6, t);
        else if (tid < 240 && t + 1 < SEQ) scatter_x(t + 1, tid - 128);
        __syncthreads();
    }
}

extern "C" void kernel_launch(void* const* d_in, const int* in_sizes, int n_in,
                              void* d_out, int out_size, void* d_ws, size_t ws_size,
                              hipStream_t stream)
{
    const float* x    = (const float*)d_in[0];
    const float* WihS = (const float*)d_in[1];
    const float* WhhS = (const float*)d_in[2];
    const float* bihS = (const float*)d_in[3];
    const float* bhhS = (const float*)d_in[4];
    const float* WihA = (const float*)d_in[5];
    const float* WhhA = (const float*)d_in[6];
    const float* bihA = (const float*)d_in[7];
    const float* bhhA = (const float*)d_in[8];
    const float* WihV = (const float*)d_in[9];
    const float* WhhV = (const float*)d_in[10];
    const float* bihV = (const float*)d_in[11];
    const float* bhhV = (const float*)d_in[12];
    const float* WihP = (const float*)d_in[13];
    const float* WhhP = (const float*)d_in[14];
    const float* bihP = (const float*)d_in[15];
    const float* bhhP = (const float*)d_in[16];
    const float* WfcA = (const float*)d_in[17];
    const float* bfcA = (const float*)d_in[18];
    const float* WfcV = (const float*)d_in[19];
    const float* bfcV = (const float*)d_in[20];
    const float* WfcP = (const float*)d_in[21];
    const float* bfcP = (const float*)d_in[22];

    __bf16* wsw = (__bf16*)d_ws;
    float* bsum = (float*)((char*)d_ws + (size_t)WSW_HALVES * sizeof(__bf16));

    prep_weights<<<(WSW_HALVES + 255) / 256, 256, 0, stream>>>(
        WihS, WhhS, WihA, WhhA, WihV, WhhV, WihP, WhhP, wsw);
    prep_bias<<<(4 * NGATE + 255) / 256, 256, 0, stream>>>(
        bihS, bhhS, bihA, bhhA, bihV, bhhV, bihP, bhhP, bsum);

    lstm_main<<<dim3(256 / BT), dim3(256), 0, stream>>>(
        x, wsw, bsum, WfcA, bfcA, WfcV, bfcV, WfcP, bfcP, (float*)d_out);
}
